// CNF_G_28089086116613
// MI455X (gfx1250) — matrix operand order repaired and checked by reading, not compiled, not run on hardware
//
#include <hip/hip_runtime.h>
#include <math.h>
#include <stdint.h>

typedef __attribute__((ext_vector_type(16))) _Float16 v16h;
typedef __attribute__((ext_vector_type(8)))  _Float16 h8;
typedef __attribute__((ext_vector_type(8)))  float    v8f;

#define BB 512
#define NN 2048
#define HH 64

// ---------------------------------------------------------------------------
// Kernel 0: one-shot f32 -> f16 conversion (8 elements / thread, b128 I/O)
// ---------------------------------------------------------------------------
__global__ __launch_bounds__(256) void cnf_cvt(const float* __restrict__ src,
                                               _Float16* __restrict__ dst, int n)
{
  const int i = (blockIdx.x * 256 + threadIdx.x) * 8;
  if (i < n) {
    float4 f0 = ((const float4*)(src + i))[0];
    float4 f1 = ((const float4*)(src + i))[1];
    h8 h;
    h[0] = (_Float16)f0.x; h[1] = (_Float16)f0.y; h[2] = (_Float16)f0.z; h[3] = (_Float16)f0.w;
    h[4] = (_Float16)f1.x; h[5] = (_Float16)f1.y; h[6] = (_Float16)f1.z; h[7] = (_Float16)f1.w;
    *(h8*)(dst + i) = h;
  }
}

// ---------------------------------------------------------------------------
// Kernel 1: scalar-t MLP -> p(320); store Wp, Up, Bp, Up*Wp into params ws.
// ---------------------------------------------------------------------------
__global__ __launch_bounds__(320) void cnf_mlp(
    const float* __restrict__ t,
    const float* __restrict__ fc1_w, const float* __restrict__ fc1_b,
    const float* __restrict__ fc2_w, const float* __restrict__ fc2_b,
    const float* __restrict__ fc3_w, const float* __restrict__ fc3_b,
    float* __restrict__ prm)   // [0:64)=Wp [64:128)=Up [128:192)=Bp [192:256)=Up*Wp
{
  __shared__ float p1[HH], p2[HH], wps[HH], ups[HH];
  const int tid = threadIdx.x;
  const float tv = t[0];

  if (tid < HH) p1[tid] = tanhf(fmaf(tv, fc1_w[tid], fc1_b[tid]));
  __syncthreads();

  if (tid < HH) {
    float s = fc2_b[tid];
#pragma unroll 8
    for (int j = 0; j < HH; ++j) s = fmaf(fc2_w[tid * HH + j], p1[j], s);
    p2[tid] = tanhf(s);
  }
  __syncthreads();

  {
    float s = fc3_b[tid];
#pragma unroll 8
    for (int j = 0; j < HH; ++j) s = fmaf(fc3_w[tid * HH + j], p2[j], s);
    if (tid < 64)                        { prm[tid] = s;              wps[tid] = s; }
    else if (tid >= 128 && tid < 192)    { prm[64 + (tid - 128)] = s; ups[tid - 128] = s; }
    else if (tid >= 256)                 { prm[128 + (tid - 256)] = s; }
  }
  __syncthreads();
  if (tid < HH) prm[192 + tid] = wps[tid] * ups[tid];
}

// ---------------------------------------------------------------------------
// Kernel 2: Zn = z @ edges^T with v_wmma_f32_16x16x32_f16.
// Block = 8 waves = 1 M-tile x 8 N-strips. The shared 16x32 A tile is staged
// through LDS with double-buffered global_load_async_to_lds_b128 (ASYNCcnt),
// read back per-wave via ds_load_b128. B fragments load straight to VGPRs.
// ---------------------------------------------------------------------------
__device__ __forceinline__ v16h ld16(const _Float16* __restrict__ p) {
  h8 lo = *(const h8*)p;        // global_load_b128
  h8 hi = *(const h8*)(p + 8);  // global_load_b128
  return __builtin_shufflevector(lo, hi, 0,1,2,3,4,5,6,7,8,9,10,11,12,13,14,15);
}

__global__ __launch_bounds__(256) void cnf_gemm(const _Float16* __restrict__ zh,
                                                const _Float16* __restrict__ eh,
                                                float* __restrict__ Zn)
{
  // Double-buffered A tile: 2 x (16 rows x 32 halves) = 2 x 1KB at LDS offset 0.
  __shared__ __align__(16) _Float16 sA[2][16 * 32];

  const int lane = threadIdx.x & 31;
  const int wv   = threadIdx.x >> 5;
  const int w    = blockIdx.x * 8 + wv;   // 0..1023 waves
  const int mt   = w >> 5;                // M tile   (0..31) — same for all 8 waves in block
  const int ns   = w & 31;                // N strip  (0..31)
  const int lr   = lane & 15;
  const int hi   = lane >> 4;

  // Async A staging addresses (used by wave 0 only): lane covers (row=lr, seg=hi),
  // two b128 per lane per tile via offset:0 / offset:32.
  const uint64_t abase = (uint64_t)(uintptr_t)(zh + (size_t)(mt * 16 + lr) * NN)
                       + (uint64_t)hi * 16;
  const uint32_t lbase = (uint32_t)(lr * 64 + hi * 16);

  // B 32x16, contiguous k order (same as A): lanes 0-15 K=0..15; lanes 16-31 K=16..31; col = lane&15
  const _Float16* b0r = eh + (size_t)(ns * 64 +  0 + lr) * NN;
  const _Float16* b1r = eh + (size_t)(ns * 64 + 16 + lr) * NN;
  const _Float16* b2r = eh + (size_t)(ns * 64 + 32 + lr) * NN;
  const _Float16* b3r = eh + (size_t)(ns * 64 + 48 + lr) * NN;

  // Prologue: stage A tile for k0 = 0 into buffer 0.
  if (wv == 0) {
    asm volatile("global_load_async_to_lds_b128 %0, %1, off"
                 :: "v"(lbase), "v"(abase) : "memory");
    asm volatile("global_load_async_to_lds_b128 %0, %1, off offset:32"
                 :: "v"(lbase), "v"(abase) : "memory");
  }
  asm volatile("s_wait_asynccnt 0x0" ::: "memory");
  __syncthreads();

  v8f c0 = {}, c1 = {}, c2 = {}, c3 = {};

  for (int k0 = 0; k0 < NN; k0 += 32) {
    const int cur = (k0 >> 5) & 1;
    const int nxt = cur ^ 1;

    // Kick off async staging of the next A tile into the other buffer.
    if (wv == 0 && (k0 + 32) < NN) {
      const uint32_t l0 = (uint32_t)(nxt * 1024) + lbase;
      const uint64_t g  = abase + (uint64_t)(k0 + 32) * 2;
      asm volatile("global_load_async_to_lds_b128 %0, %1, off"
                   :: "v"(l0), "v"(g) : "memory");
      asm volatile("global_load_async_to_lds_b128 %0, %1, off offset:32"
                   :: "v"(l0), "v"(g) : "memory");
    }
    if (k0 + 64 < NN) {
      __builtin_prefetch(b0r + k0 + 64, 0, 1);   // global_prefetch_b8
      __builtin_prefetch(b2r + k0 + 64, 0, 1);
    }

    // A fragment from LDS (this code's contiguous k order, same as B):
    //   lanes 0-15 K=0..15; lanes 16-31 K=16..31
    h8 a0 = *(const h8*)&sA[cur][lr * 32 +     hi * 16];   // ds_load_b128
    h8 a1 = *(const h8*)&sA[cur][lr * 32 + 8 + hi * 16];   // ds_load_b128
    v16h a = __builtin_shufflevector(a0, a1, 0,1,2,3,4,5,6,7,8,9,10,11,12,13,14,15);

    // All four B fragments first -> deeper load/WMMA overlap window.
    const int kb = k0 + hi * 16;
    v16h b0 = ld16(b0r + kb);
    v16h b1 = ld16(b1r + kb);
    v16h b2 = ld16(b2r + kb);
    v16h b3 = ld16(b3r + kb);

    c0 = __builtin_amdgcn_wmma_f32_16x16x32_f16(false, a, false, b0, (short)0, c0, false, false);
    c1 = __builtin_amdgcn_wmma_f32_16x16x32_f16(false, a, false, b1, (short)0, c1, false, false);
    c2 = __builtin_amdgcn_wmma_f32_16x16x32_f16(false, a, false, b2, (short)0, c2, false, false);
    c3 = __builtin_amdgcn_wmma_f32_16x16x32_f16(false, a, false, b3, (short)0, c3, false, false);

    // Next tile staged + everyone done reading the current buffer.
    asm volatile("s_wait_asynccnt 0x0" ::: "memory");
    __syncthreads();
  }

  // C/D layout: VGPR r -> M = r + hi*8, N = lane&15
  float* orow = Zn + (size_t)(mt * 16 + hi * 8) * NN + ns * 64 + lr;
#pragma unroll
  for (int r = 0; r < 8; ++r) {
    orow[(size_t)r * NN +  0] = c0[r];
    orow[(size_t)r * NN + 16] = c1[r];
    orow[(size_t)r * NN + 32] = c2[r];
    orow[(size_t)r * NN + 48] = c3[r];
  }
}

// ---------------------------------------------------------------------------
// Kernel 3: epilogue. Per (b,n): 64-wide tanh gate reduction.
// ---------------------------------------------------------------------------
__global__ __launch_bounds__(256) void cnf_epilogue(const float* __restrict__ Zn,
                                                    const float* __restrict__ edges,
                                                    const float* __restrict__ prm,
                                                    float* __restrict__ out)
{
  __shared__ float sW[HH], sU[HH], sB[HH], sUW[HH];
  __shared__ float red[256];
  const int tid = threadIdx.x;
  const int b   = blockIdx.x;

  if (tid < HH) {
    sW[tid]  = prm[tid];
    sU[tid]  = prm[64 + tid];
    sB[tid]  = prm[128 + tid];
    sUW[tid] = prm[192 + tid];
  }
  __syncthreads();

  float rsum = 0.0f;
  for (int n = tid; n < NN; n += 256) {
    const float zn = Zn[(size_t)b * NN + n];
    float accH = 0.0f, accJ = 0.0f;
#pragma unroll 8
    for (int k = 0; k < HH; ++k) {
      const float h = tanhf(fmaf(zn, sW[k], sB[k]));
      accH = fmaf(h, sU[k], accH);
      accJ = fmaf(sUW[k], fmaf(-h, h, 1.0f), accJ);
    }
    out[(size_t)b * NN + n] = accH * (1.0f / HH);
    rsum = fmaf(accJ * (1.0f / HH), edges[(size_t)n * NN + n], rsum);
  }

  red[tid] = rsum;
  __syncthreads();
  for (int s = 128; s > 0; s >>= 1) {
    if (tid < s) red[tid] += red[tid + s];
    __syncthreads();
  }
  if (tid == 0) out[(size_t)BB * NN + b] = -red[0];
}

// ---------------------------------------------------------------------------
extern "C" void kernel_launch(void* const* d_in, const int* in_sizes, int n_in,
                              void* d_out, int out_size, void* d_ws, size_t ws_size,
                              hipStream_t stream)
{
  const float* t     = (const float*)d_in[0];
  const float* z     = (const float*)d_in[1];
  const float* edges = (const float*)d_in[2];
  const float* fc1_w = (const float*)d_in[3];
  const float* fc1_b = (const float*)d_in[4];
  const float* fc2_w = (const float*)d_in[5];
  const float* fc2_b = (const float*)d_in[6];
  const float* fc3_w = (const float*)d_in[7];
  const float* fc3_b = (const float*)d_in[8];

  char* wsb = (char*)d_ws;
  float*    Zn  = (float*)   (wsb);                                    // 4 MB
  float*    prm = (float*)   (wsb + (size_t)BB * NN * 4);              // 1 KB
  _Float16* zh  = (_Float16*)(wsb + (size_t)BB * NN * 4 + 1024);       // 2 MB
  _Float16* eh  = (_Float16*)(wsb + (size_t)BB * NN * 4 + 1024
                                  + (size_t)BB * NN * 2);              // 8 MB

  cnf_cvt<<<(BB * NN) / (256 * 8), 256, 0, stream>>>(z, zh, BB * NN);
  cnf_cvt<<<(NN * NN) / (256 * 8), 256, 0, stream>>>(edges, eh, NN * NN);
  cnf_mlp<<<1, 320, 0, stream>>>(t, fc1_w, fc1_b, fc2_w, fc2_b, fc3_w, fc3_b, prm);
  cnf_gemm<<<128, 256, 0, stream>>>(zh, eh, Zn);
  cnf_epilogue<<<BB, 256, 0, stream>>>(Zn, edges, prm, (float*)d_out);
}
